// PredModel_43465069035810
// MI455X (gfx1250) — matrix operand order repaired and checked by reading, not compiled, not run on hardware
//
#include <hip/hip_runtime.h>

typedef _Float16 h16;
typedef __attribute__((ext_vector_type(16))) _Float16 v16h;
typedef __attribute__((ext_vector_type(8)))  float    v8f;
typedef __attribute__((ext_vector_type(4)))  float    f32x4;

#define B_ 4096
#define T_ 512
#define I_ 15
#define H_ 64
#define LDSPITCH 72   // 64 + 8 halves pad -> conflict-free b128 loads

struct F4x2 { f32x4 a, b; };

__device__ __forceinline__ float fast_tanh(float x) {
#if __has_builtin(__builtin_amdgcn_tanhf)
  return __builtin_amdgcn_tanhf(x);      // gfx1250 v_tanh_f32 (1 trans op)
#else
  float e = __builtin_amdgcn_exp2f(x * 2.8853900817779268f);
  return 1.0f - 2.0f * __builtin_amdgcn_rcpf(e + 1.0f);
#endif
}

// LDS-only workgroup barrier: make our DS stores visible, sync, and do NOT
// drain vmem loadcnt (unlike __syncthreads(), whose global fence serializes
// the in-flight pipelined x loads with the barrier every timestep).
__device__ __forceinline__ void wg_barrier_signal_lds() {
  asm volatile("s_wait_dscnt 0x0\n\t"
               "s_barrier_signal -1" ::: "memory");
}
__device__ __forceinline__ void wg_barrier_wait_lds() {
  asm volatile("s_barrier_wait -1" ::: "memory");
}

__launch_bounds__(256, 2)
__global__ void rnn_fused_kernel(const float* __restrict__ x,
                                 const float* __restrict__ W_ih,
                                 const float* __restrict__ W_hh,
                                 const float* __restrict__ b_ih,
                                 const float* __restrict__ b_hh,
                                 const float* __restrict__ fc_w,
                                 const float* __restrict__ fc_b,
                                 float* __restrict__ out) {
  // ping-pong h buffers: [parity][row-tile][16 rows * padded pitch]
  __shared__ h16 hbuf[2][2][16 * LDSPITCH];

  const int tid  = threadIdx.x;
  const int wave = tid >> 5;
  const int lane = tid & 31;
  const int r    = wave & 1;   // row-tile (16 batch rows) within block
  const int s    = wave >> 1;  // n-tile: this wave computes n in [16s, 16s+16)
  const int lh   = lane >> 4;  // lane half
  const int ln   = lane & 15;

  // ---------- Build resident B-fragments (16x16x32 f16 layout) ----------
  // B layout (K x N = 32x16): halves j -> k = 16*(j/8) + lh*8 + (j%8),
  // n = ln. C/D layout: lane l, vgpr i -> (m = lh*8 + i, n = ln).
  v16h wb[2];   // [k-half] of W_hh^T for this wave's n-tile  (B[k][n] = W_hh[n][k])
  v16h wib;     // input-proj B frag; K padded 15 -> 32, bias folded at k=15
  {
    const int n = s * 16 + ln;
    const float bias_n = b_ih[n] + b_hh[n];
    #pragma unroll
    for (int kh = 0; kh < 2; ++kh)
      #pragma unroll
      for (int j = 0; j < 16; ++j) {
        const int k = kh * 32 + lh * 8 + (j & 7) + ((j >> 3) << 4);
        wb[kh][j] = (h16)W_hh[n * H_ + k];
      }
    #pragma unroll
    for (int j = 0; j < 16; ++j) {
      const int k = lh * 8 + (j & 7) + ((j >> 3) << 4);
      // k < 15: real W_ih column; k == 15: bias column (A supplies 1.0 there)
      wib[j] = (k < I_) ? (h16)W_ih[n * I_ + k]
                        : ((k == I_) ? (h16)bias_n : (h16)0.0f);
    }
  }

  // ---------- x streaming (A-frag 16x32: lane l holds m=ln, k=16*(j/8)+8*lh+(j%8)) ----
  const int row   = blockIdx.x * 32 + r * 16 + ln;
  const int koff  = lh * 8;                // real k for j=0..7; j=8..15 stay zero
  const int i7off = 7 - lh;                // lh=0: k=7 (used); lh=1: k=14 (discarded, in-bounds)
  int       xidx  = row * T_ * I_ + koff;  // base for current timestep t
  const h16 a7fix = (h16)1.0f;             // k=15 slot: 1.0 -> multiplies bias column

  // prologue: load raw x floats for t=0 (consumed at first iteration top)
  float cur[8];
  #pragma unroll
  for (int j = 0; j < 7; ++j) cur[j] = x[xidx + j];
  cur[7] = x[xidx + i7off];

  v16h hA0 = {}, hA1 = {};                 // h(t-1) A-fragments, h0 = 0

  #pragma unroll 2                         // constant ping-pong parity, mov-free rotation
  for (int t = 0; t < T_; ++t) {
    // ---- build A fragment for step t from pipelined registers (no mem wait here)
    v16h ax = {};
    #pragma unroll
    for (int j = 0; j < 7; ++j) ax[j] = (h16)cur[j];
    ax[7] = lh ? a7fix : (h16)cur[7];      // lh=1: k=15 -> 1.0 (bias); lh=0: k=7

    // ---- issue loads for t+1 now; they complete across the barrier
    const int bn = (t < T_ - 1) ? (xidx + I_) : xidx;   // uniform select at tail
    float nxt[8];
    #pragma unroll
    for (int j = 0; j < 7; ++j) nxt[j] = x[bn + j];
    nxt[7] = x[bn + i7off];

    // ---- 3 WMMAs, accumulator starts at inline 0 (bias comes via k=15 column)
    v8f acc = {};
    acc = __builtin_amdgcn_wmma_f32_16x16x32_f16(false, ax,  false, wib,   (short)0, acc, false, false);
    acc = __builtin_amdgcn_wmma_f32_16x16x32_f16(false, hA0, false, wb[0], (short)0, acc, false, false);
    acc = __builtin_amdgcn_wmma_f32_16x16x32_f16(false, hA1, false, wb[1], (short)0, acc, false, false);

    // ---- tanh, f16 convert, scatter into ping-pong LDS h buffer
    const int p = t & 1;
    h16* hb = &hbuf[p][r][0];
    #pragma unroll
    for (int i = 0; i < 8; ++i) {
      const int m = lh * 8 + i;
      hb[m * LDSPITCH + s * 16 + ln] = (h16)fast_tanh(acc[i]);
    }

    // ---- LDS-only split barrier; prefetch rides in the signal->wait gap
    wg_barrier_signal_lds();
    __builtin_prefetch(&x[xidx + 4 * I_], 0, 3);   // ~4 steps (~240B) ahead -> GL2
    wg_barrier_wait_lds();

    // ---- reload h(t) as two A-fragments (4x ds_load_b128, conflict-free)
    const h16* hr = &hbuf[p][r][ln * LDSPITCH];
    F4x2 p0, p1;
    p0.a = *(const f32x4*)(hr + (     lh * 8));   // k =      lh*8 .. +7  (j 0..7)
    p0.b = *(const f32x4*)(hr + (16 + lh * 8));   // k = 16 + lh*8 .. +7  (j 8..15)
    p1.a = *(const f32x4*)(hr + (32 + lh * 8));
    p1.b = *(const f32x4*)(hr + (48 + lh * 8));
    hA0 = __builtin_bit_cast(v16h, p0);
    hA1 = __builtin_bit_cast(v16h, p1);

    // ---- rotate x pipeline
    #pragma unroll
    for (int j = 0; j < 8; ++j) cur[j] = nxt[j];
    xidx = bn;
  }

  // ---------- head: out[b] = h_T . fc_w + fc_b (32 rows per block) ----------
  if (tid < 32) {
    const int rr = tid >> 4, mm = tid & 15;
    const h16* hr = &hbuf[(T_ - 1) & 1][rr][mm * LDSPITCH];
    float sum = fc_b[0];
    #pragma unroll
    for (int n = 0; n < H_; ++n) sum += (float)hr[n] * fc_w[n];
    out[blockIdx.x * 32 + tid] = sum;
  }
}

extern "C" void kernel_launch(void* const* d_in, const int* in_sizes, int n_in,
                              void* d_out, int out_size, void* d_ws, size_t ws_size,
                              hipStream_t stream) {
  const float* x    = (const float*)d_in[0];
  const float* W_ih = (const float*)d_in[1];
  const float* W_hh = (const float*)d_in[2];
  const float* b_ih = (const float*)d_in[3];
  const float* b_hh = (const float*)d_in[4];
  const float* fc_w = (const float*)d_in[5];
  const float* fc_b = (const float*)d_in[6];

  dim3 grid(B_ / 32);   // 128 blocks x 8 waves = 1024 waves
  dim3 block(256);
  rnn_fused_kernel<<<grid, block, 0, stream>>>(x, W_ih, W_hh, b_ih, b_hh,
                                               fc_w, fc_b, (float*)d_out);
}
